// ProxyGraphModel_6098853560870
// MI455X (gfx1250) — matrix operand order repaired and checked by reading, not compiled, not run on hardware
//
#include <hip/hip_runtime.h>
#include <stdint.h>

typedef _Float16 h16;
typedef __attribute__((ext_vector_type(16))) _Float16 v16h;
typedef __attribute__((ext_vector_type(8)))  float    v8f;

#define NB    256   // batch
#define NEL   90
#define NMX   270
#define PIN   448
#define NH    3
#define EPSB  1e-5f

union FragH { v16h v; uint4 q[2]; };
union Cvt32 { uint32_t u; h16 hh[2]; };
union Cvt64 { uint2 u; h16 hh[4]; };

// ---------------------------------------------------------------------------
// small prep kernels
// ---------------------------------------------------------------------------

__global__ void prep_latsg(const float* __restrict__ lat, const int* __restrict__ sg,
                           const float* W1, const float* b1,
                           const float* g1, const float* be1, const float* m1, const float* v1,
                           const float* W2, const float* b2,
                           const float* g2, const float* be2, const float* m2, const float* v2,
                           const float* sgtab, float* latx, float* sgx)
{
    int b = blockIdx.x, c = threadIdx.x;     // 128 threads
    __shared__ float sh[128];
    float x = b1[c];
    for (int k = 0; k < 6; k++) x += lat[b*6 + k] * W1[k*128 + c];
    x = (x - m1[c]) * g1[c] * rsqrtf(v1[c] + EPSB) + be1[c];
    sh[c] = x;
    __syncthreads();
    float y = b2[c];
    for (int k = 0; k < 128; k++) y += sh[k] * W2[k*128 + c];
    y = (y - m2[c]) * g2[c] * rsqrtf(v2[c] + EPSB) + be2[c];
    latx[b*128 + c] = y;
    if (c < 64) sgx[b*64 + c] = sgtab[sg[b]*64 + c];
}

__global__ void calc_z(const int* __restrict__ comp, int* __restrict__ z, int* __restrict__ natoms)
{
    int b = blockIdx.x, tid = threadIdx.x;   // 128 threads
    __shared__ int cum[NEL];
    if (tid == 0) {
        int s = 0;
        for (int e = 0; e < NEL; e++) { s += comp[b*NEL + e]; cum[e] = s; }
        natoms[b] = s;
    }
    __syncthreads();
    int na = cum[NEL-1];
    for (int t = tid; t < NMX; t += 128) {
        int zi = 0;
        if (t < na) { for (int e = 0; e < NEL; e++) zi += (cum[e] <= t); }
        z[b*NMX + t] = zi;
    }
}

__global__ void calc_physE(const float* z_emb, const float* ptab, const float* gtab,
                           const int* pidx, const int* gidx,
                           const float* props, const float* Wp, const float* bp,
                           const float* Wphys, const float* bphys, float* physE)
{
    int e = blockIdx.x, c = threadIdx.x;     // 256 threads
    __shared__ float pin[128];
    if (c < 32)       pin[c] = z_emb[e*32 + c];
    else if (c < 64)  pin[c] = ptab[pidx[e]*32 + (c-32)];
    else if (c < 96)  pin[c] = gtab[gidx[e]*32 + (c-64)];
    else if (c < 128) {
        int pp = c - 96;
        float s = bp[pp];
        for (int k = 0; k < 20; k++) s += props[e*20 + k] * Wp[k*32 + pp];
        pin[c] = s;
    }
    __syncthreads();
    float s = bphys[c];
    for (int k = 0; k < 128; k++) s += pin[k] * Wphys[k*256 + c];
    physE[e*256 + c] = s;
}

__global__ void build_node(const float* __restrict__ physE, const float* __restrict__ latx,
                           const float* __restrict__ sgx, const int* __restrict__ z,
                           const int* __restrict__ natoms, h16* __restrict__ node)
{
    int i = blockIdx.x, b = blockIdx.y, c = threadIdx.x;   // 448 threads
    int zi = z[b*NMX + i];
    float v;
    if (c < 256)      v = physE[zi*256 + c];
    else if (c < 384) v = latx[b*128 + (c-256)];
    else              v = sgx[b*64 + (c-384)];
    if (i >= natoms[b]) v = 0.f;
    node[((long)(b*NMX + i))*PIN + c] = (h16)v;
}

__global__ void f32_to_f16(const float* __restrict__ x, h16* __restrict__ y, long n)
{
    long i = (long)blockIdx.x * blockDim.x + threadIdx.x;
    if (i < n) y[i] = (h16)x[i];
}

__global__ void bn_lrelu(const float* __restrict__ x, const float* g, const float* be,
                         const float* m, const float* v, h16* __restrict__ y, int C, long total)
{
    long idx = (long)blockIdx.x * blockDim.x + threadIdx.x;
    if (idx < total) {
        int c = (int)(idx % C);
        float t = (x[idx] - m[c]) * g[c] * rsqrtf(v[c] + EPSB) + be[c];
        t = (t >= 0.f) ? t : 0.2f * t;
        y[idx] = (h16)t;
    }
}

// ---------------------------------------------------------------------------
// generic f16 WMMA GEMM: C[M,N](f32 and/or f16) = A[M,K]h * B[K,N]h (+bias)
// block tile 64x128, 8 waves (2m x 4n), each wave 2x2 wmma tiles, K step 32.
// M % 64 == 0, K % 32 == 0; N arbitrary (uniform fast path when the block's
// 128 columns are fully in range, branchy edge path otherwise).
// A staged row-major [r][k] (stride 40h = 80B, 16B aligned rows).
// B staged TRANSPOSED [n][k] (stride 40h) so each B fragment is 2x b128 reads.
// ---------------------------------------------------------------------------
#define GAP 40    // As row stride (halves), 80B: multiple of 16B
#define GBP 40    // Bs (transposed) row stride (halves)

__global__ __launch_bounds__(256)
void gemm_f16(const h16* __restrict__ A, const h16* __restrict__ Bm,
              float* Cf, h16* Ch, const float* bias,
              int M, int N, int K, int lda, int ldb, int ldc)
{
    __shared__ __align__(16) h16 As[64 * GAP];
    __shared__ __align__(16) h16 Bs[128 * GBP];
    int tid = threadIdx.x;
    int lane = tid & 31, w = tid >> 5;
    int l15 = lane & 15, hi = lane >> 4;
    int wm = w & 1, wn = w >> 1;
    long mBase = (long)blockIdx.y * 64;
    int  nBase = blockIdx.x * 128;
    const bool fullN = (nBase + 128 <= N);     // uniform across block

    v8f acc[2][2];
    v8f zero = {0.f,0.f,0.f,0.f,0.f,0.f,0.f,0.f};
    for (int a = 0; a < 2; a++) for (int b = 0; b < 2; b++) acc[a][b] = zero;

    int arow = tid >> 2, aseg = tid & 3;       // A staging: 1 uint4 per thread
    int bkk = tid >> 3, bnp = (tid & 7) * 8;   // B staging: 8 cols x 1 row per thread

    for (int k0 = 0; k0 < K; k0 += 32) {
        __syncthreads();
        // stage A tile: 64 rows x 32 halves, one b128 per thread
        {
            const uint4* src = reinterpret_cast<const uint4*>(
                A + (mBase + arow)*(long)lda + k0 + aseg*8);
            *reinterpret_cast<uint4*>(&As[arow*GAP + aseg*8]) = *src;
        }
        // stage B tile transposed [n][k]
        if (fullN) {
            // fast path: thread covers 8 consecutive columns at one k (4x u32)
            const h16* brow = Bm + (long)(k0 + bkk)*ldb + nBase + bnp;
            #pragma unroll
            for (int q = 0; q < 4; q++) {
                Cvt32 cv;
                cv.u = *reinterpret_cast<const uint32_t*>(brow + 2*q);
                Bs[(bnp + 2*q + 0)*GBP + bkk] = cv.hh[0];
                Bs[(bnp + 2*q + 1)*GBP + bkk] = cv.hh[1];
            }
        } else {
            for (int idx = tid; idx < 32*64; idx += 256) {
                int kk = idx >> 6, np = idx & 63;
                int n0 = nBase + 2*np;
                Cvt32 cv; cv.u = 0;
                if (n0 + 1 < N) {
                    cv.u = *reinterpret_cast<const uint32_t*>(Bm + (long)(k0+kk)*ldb + n0);
                } else if (n0 < N) {
                    h16 hv = Bm[(long)(k0+kk)*ldb + n0];
                    cv.u = (uint32_t)*reinterpret_cast<uint16_t*>(&hv);
                }
                int nl = 2*np;
                Bs[(nl+0)*GBP + kk] = cv.hh[0];
                Bs[(nl+1)*GBP + kk] = cv.hh[1];
            }
        }
        // prefetch next K tiles while this one is consumed
        if (k0 + 32 < K) {
            __builtin_prefetch(A + (mBase + arow)*(long)lda + k0 + 32, 0, 1);
            __builtin_prefetch(Bm + (long)(k0 + 32 + (tid & 31))*ldb + nBase, 0, 1);
        }
        __syncthreads();

        FragH af[2], bf[2];
        for (int mi = 0; mi < 2; mi++) {
            int r = wm*32 + mi*16 + l15;
            const h16* ap = &As[r*GAP + hi*8];
            af[mi].q[0] = *reinterpret_cast<const uint4*>(ap);        // K = hi*8 + 0..7
            af[mi].q[1] = *reinterpret_cast<const uint4*>(ap + 16);   // K = 16 + hi*8 + 0..7
        }
        for (int ni = 0; ni < 2; ni++) {
            int n = wn*32 + ni*16 + l15;
            const h16* bp = &Bs[n*GBP + hi*8];
            bf[ni].q[0] = *reinterpret_cast<const uint4*>(bp);        // K = hi*8 + 0..7
            bf[ni].q[1] = *reinterpret_cast<const uint4*>(bp + 16);   // K = 16 + hi*8 + 0..7
        }
        for (int mi = 0; mi < 2; mi++)
            for (int ni = 0; ni < 2; ni++)
                acc[mi][ni] = __builtin_amdgcn_wmma_f32_16x16x32_f16(
                    false, af[mi].v, false, bf[ni].v, (short)0, acc[mi][ni], false, false);
    }

    for (int mi = 0; mi < 2; mi++)
        for (int ni = 0; ni < 2; ni++) {
            long row0 = mBase + wm*32 + mi*16;
            int col = nBase + wn*32 + ni*16 + l15;
            if (col < N) {
                float bv = bias ? bias[col] : 0.f;
                for (int r = 0; r < 8; r++) {
                    long row = row0 + r + hi*8;
                    float v = acc[mi][ni][r] + bv;
                    if (Cf) Cf[row*(long)ldc + col] = v;
                    if (Ch) Ch[row*(long)ldc + col] = (h16)v;
                }
            }
        }
}

// ---------------------------------------------------------------------------
// attention logits: asrc/adst[b,i,h] = sum_c h[b,i,h,c] * a_{s,d}[h,c]
// ---------------------------------------------------------------------------
__global__ void att_logits(const h16* __restrict__ Hm, const float* __restrict__ a_s,
                           const float* __restrict__ a_d,
                           float* __restrict__ asrc, float* __restrict__ adst, int C, int ldh)
{
    int i = blockIdx.x, b = blockIdx.y, tid = threadIdx.x;  // 128 threads
    __shared__ float rs[128], rd[128];
    long rowoff = ((long)(b*NMX + i)) * ldh;
    for (int h = 0; h < NH; h++) {
        float sa = 0.f, sd = 0.f;
        for (int c = tid; c < C; c += 128) {
            float hv = (float)Hm[rowoff + h*C + c];
            sa += hv * a_s[h*C + c];
            sd += hv * a_d[h*C + c];
        }
        rs[tid] = sa; rd[tid] = sd;
        __syncthreads();
        for (int s = 64; s > 0; s >>= 1) {
            if (tid < s) { rs[tid] += rs[tid+s]; rd[tid] += rd[tid+s]; }
            __syncthreads();
        }
        if (tid == 0) {
            long o = ((long)(b*NMX + i))*NH + h;
            asrc[o] = rs[0]; adst[o] = rd[0];
        }
        __syncthreads();
    }
}

// row softmax stats over valid sources j
__global__ void att_rowstats(const float* __restrict__ asrc, const float* __restrict__ adst,
                             const int* __restrict__ natoms,
                             float* __restrict__ rmax, float* __restrict__ rsum)
{
    int i = blockIdx.x, b = blockIdx.y, tid = threadIdx.x;  // 128 threads
    __shared__ float red[128];
    int na = natoms[b];
    for (int h = 0; h < NH; h++) {
        long base = (long)(b*NMX)*NH + h;
        float adi = adst[(long)(b*NMX + i)*NH + h];
        float mx = -1e30f;
        for (int j = tid; j < NMX; j += 128) {
            bool valid = ((i < na) && (j < na)) || (j == i);
            if (valid) {
                float l = asrc[base + (long)j*NH] + adi;
                l = (l >= 0.f) ? l : 0.2f * l;
                mx = fmaxf(mx, l);
            }
        }
        red[tid] = mx; __syncthreads();
        for (int s = 64; s > 0; s >>= 1) { if (tid < s) red[tid] = fmaxf(red[tid], red[tid+s]); __syncthreads(); }
        mx = red[0]; __syncthreads();
        float sm = 0.f;
        for (int j = tid; j < NMX; j += 128) {
            bool valid = ((i < na) && (j < na)) || (j == i);
            if (valid) {
                float l = asrc[base + (long)j*NH] + adi;
                l = (l >= 0.f) ? l : 0.2f * l;
                sm += __expf(l - mx);
            }
        }
        red[tid] = sm; __syncthreads();
        for (int s = 64; s > 0; s >>= 1) { if (tid < s) red[tid] += red[tid+s]; __syncthreads(); }
        if (tid == 0) {
            long o = (long)(b*NMX + i)*NH + h;
            rmax[o] = mx; rsum[o] = red[0];
        }
        __syncthreads();
    }
}

// ---------------------------------------------------------------------------
// GAT aggregation: X[b,i,c] = (1/3) * sum_h sum_j alpha(b,i,j,h)*H[b,j,h,c] + bias[c]
// alpha built on the fly in A-fragments. block tile 64(i) x 64(c), 8 waves (4m x 2n).
// H tile staged TRANSPOSED [c][j] in LDS so B fragments are 2x b128 reads.
// ---------------------------------------------------------------------------
#define ABP 40

__global__ __launch_bounds__(256)
void gat_aggregate(const h16* __restrict__ Hm, const float* __restrict__ asrc,
                   const float* __restrict__ adst, const float* __restrict__ rmax,
                   const float* __restrict__ rsum, const int* __restrict__ natoms,
                   const float* __restrict__ bias,
                   float* Xf, h16* Xh, int C, int ldh)
{
    __shared__ float s_asrc[288];
    __shared__ __align__(16) h16 Bs[64 * ABP];
    int b = blockIdx.z;
    int iBase = blockIdx.y * 64;
    int cBase = blockIdx.x * 64;
    int tid = threadIdx.x, lane = tid & 31, w = tid >> 5;
    int l15 = lane & 15, hi = lane >> 4;
    int wm = w & 3, wn = w >> 2;
    int na = natoms[b];
    int i = iBase + wm*16 + l15;
    bool irow = (i < NMX);

    v8f acc[2];
    v8f zero = {0.f,0.f,0.f,0.f,0.f,0.f,0.f,0.f};
    acc[0] = zero; acc[1] = zero;

    for (int h = 0; h < NH; h++) {
        __syncthreads();
        for (int j = tid; j < NMX; j += 256)
            s_asrc[j] = asrc[((long)(b*NMX + j))*NH + h];
        __syncthreads();

        float adi = 0.f, rmi = 0.f, inv_rs = 0.f;
        if (irow) {
            long o = ((long)(b*NMX + i))*NH + h;
            adi = adst[o]; rmi = rmax[o]; inv_rs = 1.f / rsum[o];
        }

        for (int j0 = 0; j0 < 288; j0 += 32) {
            __syncthreads();
            // stage H tile transposed [c][j]: uint2 covers 4 channels at one j
            for (int idx = tid; idx < 512; idx += 256) {
                int kk = idx >> 4, c4 = (idx & 15) * 4;
                int j = j0 + kk;
                Cvt64 cv; cv.u.x = 0; cv.u.y = 0;
                if (j < NMX)
                    cv.u = *reinterpret_cast<const uint2*>(
                        Hm + ((long)(b*NMX + j))*ldh + h*C + cBase + c4);
                Bs[(c4+0)*ABP + kk] = cv.hh[0];
                Bs[(c4+1)*ABP + kk] = cv.hh[1];
                Bs[(c4+2)*ABP + kk] = cv.hh[2];
                Bs[(c4+3)*ABP + kk] = cv.hh[3];
            }
            __syncthreads();

            v16h af;
            for (int t = 0; t < 16; t++) {
                int kk = ((t < 8) ? t : t + 8) + hi*8;
                int j = j0 + kk;
                float a = 0.f;
                if (irow && j < NMX) {
                    bool valid = ((i < na) && (j < na)) || (j == i);
                    if (valid) {
                        float l = s_asrc[j] + adi;
                        l = (l >= 0.f) ? l : 0.2f * l;
                        a = __expf(l - rmi) * inv_rs;
                    }
                }
                af[t] = (h16)a;
            }
            for (int ni = 0; ni < 2; ni++) {
                FragH fb;
                int n = wn*32 + ni*16 + l15;
                const h16* bp = &Bs[n*ABP + hi*8];
                fb.q[0] = *reinterpret_cast<const uint4*>(bp);
                fb.q[1] = *reinterpret_cast<const uint4*>(bp + 16);
                acc[ni] = __builtin_amdgcn_wmma_f32_16x16x32_f16(
                    false, af, false, fb.v, (short)0, acc[ni], false, false);
            }
        }
    }

    for (int ni = 0; ni < 2; ni++) {
        int col = cBase + wn*32 + ni*16 + l15;
        float bv = bias[col];
        for (int r = 0; r < 8; r++) {
            long row = iBase + wm*16 + r + hi*8;
            if (row < NMX) {
                float v = acc[ni][r] * (1.f/3.f) + bv;
                long off = ((long)b*NMX + row)*(long)C + col;
                if (Xf) Xf[off] = v;
                if (Xh) Xh[off] = (h16)v;
            }
        }
    }
}

// comp_x = sum_i x2 ; feat = [comp_x, sg_x, lat_x] (f16)
__global__ void sum_feat(const float* __restrict__ x2, const float* __restrict__ sgx,
                         const float* __restrict__ latx, h16* __restrict__ feat)
{
    int b = blockIdx.x, c = threadIdx.x;   // 256 threads
    float s = 0.f;
    for (int i = 0; i < NMX; i++) s += x2[((long)(b*NMX + i))*256 + c];
    feat[b*PIN + c] = (h16)s;
    if (c < 64)  feat[b*PIN + 256 + c] = (h16)sgx[b*64 + c];
    if (c < 128) feat[b*PIN + 320 + c] = (h16)latx[b*128 + c];
}

// ---------------------------------------------------------------------------
extern "C" void kernel_launch(void* const* d_in, const int* in_sizes, int n_in,
                              void* d_out, int out_size, void* d_ws, size_t ws_size,
                              hipStream_t stream)
{
    (void)in_sizes; (void)n_in; (void)ws_size;
    const int*   comp   = (const int*)  d_in[0];
    const int*   sg     = (const int*)  d_in[1];
    const float* lat    = (const float*)d_in[2];
    const int*   pidx   = (const int*)  d_in[3];
    const int*   gidx   = (const int*)  d_in[4];
    const float* sgtab  = (const float*)d_in[5];
    const float* lW1    = (const float*)d_in[6];
    const float* lb1    = (const float*)d_in[7];
    const float* lg1    = (const float*)d_in[8];
    const float* lbe1   = (const float*)d_in[9];
    const float* lm1    = (const float*)d_in[10];
    const float* lv1    = (const float*)d_in[11];
    const float* lW2    = (const float*)d_in[12];
    const float* lb2    = (const float*)d_in[13];
    const float* lg2    = (const float*)d_in[14];
    const float* lbe2   = (const float*)d_in[15];
    const float* lm2    = (const float*)d_in[16];
    const float* lv2    = (const float*)d_in[17];
    const float* z_emb  = (const float*)d_in[18];
    const float* ptab   = (const float*)d_in[19];
    const float* gtab   = (const float*)d_in[20];
    const float* props  = (const float*)d_in[21];
    const float* Wp     = (const float*)d_in[22];
    const float* bp     = (const float*)d_in[23];
    const float* Wphys  = (const float*)d_in[24];
    const float* bphys  = (const float*)d_in[25];
    const float* g1W    = (const float*)d_in[26];
    const float* g1as   = (const float*)d_in[27];
    const float* g1ad   = (const float*)d_in[28];
    const float* g1b    = (const float*)d_in[29];
    const float* g2W    = (const float*)d_in[30];
    const float* g2as   = (const float*)d_in[31];
    const float* g2ad   = (const float*)d_in[32];
    const float* g2b    = (const float*)d_in[33];
    const float* pW1    = (const float*)d_in[34];
    const float* pb1    = (const float*)d_in[35];
    const float* pbn1g  = (const float*)d_in[36];
    const float* pbn1b  = (const float*)d_in[37];
    const float* pbn1m  = (const float*)d_in[38];
    const float* pbn1v  = (const float*)d_in[39];
    const float* pW2    = (const float*)d_in[40];
    const float* pb2    = (const float*)d_in[41];
    const float* pbn2g  = (const float*)d_in[42];
    const float* pbn2b  = (const float*)d_in[43];
    const float* pbn2m  = (const float*)d_in[44];
    const float* pbn2v  = (const float*)d_in[45];
    const float* pW3    = (const float*)d_in[46];
    const float* pb3    = (const float*)d_in[47];
    float* out = (float*)d_out; (void)out_size;

    char* ws = (char*)d_ws;
    size_t off = 0;
    auto take = [&](size_t bytes) -> char* {
        char* p = ws + off;
        off += (bytes + 255) & ~(size_t)255;
        return p;
    };
    const long NR = (long)NB * NMX;           // 69120 rows

    float* sgx   = (float*)take((size_t)NB*64*4);
    float* latx  = (float*)take((size_t)NB*128*4);
    float* physE = (float*)take((size_t)NEL*256*4);
    int*   zbuf  = (int*)  take((size_t)NB*NMX*4);
    int*   nat   = (int*)  take((size_t)NB*4);
    h16* g1Wh = (h16*)take((size_t)448*1344*2);
    h16* g2Wh = (h16*)take((size_t)448*768*2);
    h16* pW1h = (h16*)take((size_t)448*512*2);
    h16* pW2h = (h16*)take((size_t)512*512*2);
    h16* pW3h = (h16*)take((size_t)512*2);
    float* as1 = (float*)take((size_t)NR*NH*4);
    float* ad1 = (float*)take((size_t)NR*NH*4);
    float* rm1 = (float*)take((size_t)NR*NH*4);
    float* rs1 = (float*)take((size_t)NR*NH*4);
    float* as2 = (float*)take((size_t)NR*NH*4);
    float* ad2 = (float*)take((size_t)NR*NH*4);
    float* rm2 = (float*)take((size_t)NR*NH*4);
    float* rs2 = (float*)take((size_t)NR*NH*4);
    h16* x1h   = (h16*)take((size_t)NR*448*2);
    h16* feath = (h16*)take((size_t)NB*PIN*2);
    float* hb1 = (float*)take((size_t)NB*512*4);
    h16*  hb1h = (h16*) take((size_t)NB*512*2);
    float* hb2 = (float*)take((size_t)NB*512*4);
    h16*  hb2h = (h16*) take((size_t)NB*512*2);
    // aliased regions
    char* regNX = take((size_t)NR*256*4);                  // max(node 62MB, x2f 71MB)
    char* regH  = take((size_t)NR*1344*2);                 // max(h1 186MB, h2 106MB)
    h16*   node = (h16*)regNX;
    float* x2f  = (float*)regNX;
    h16*   h1   = (h16*)regH;
    h16*   h2   = (h16*)regH;

    // prep
    prep_latsg<<<NB, 128, 0, stream>>>(lat, sg, lW1, lb1, lg1, lbe1, lm1, lv1,
                                       lW2, lb2, lg2, lbe2, lm2, lv2, sgtab, latx, sgx);
    calc_z<<<NB, 128, 0, stream>>>(comp, zbuf, nat);
    calc_physE<<<NEL, 256, 0, stream>>>(z_emb, ptab, gtab, pidx, gidx, props, Wp, bp,
                                        Wphys, bphys, physE);
    build_node<<<dim3(NMX, NB), 448, 0, stream>>>(physE, latx, sgx, zbuf, nat, node);

    // weight conversions to f16
    f32_to_f16<<<(448*1344 + 255)/256, 256, 0, stream>>>(g1W, g1Wh, (long)448*1344);
    f32_to_f16<<<(448*768  + 255)/256, 256, 0, stream>>>(g2W, g2Wh, (long)448*768);
    f32_to_f16<<<(448*512  + 255)/256, 256, 0, stream>>>(pW1, pW1h, (long)448*512);
    f32_to_f16<<<(512*512  + 255)/256, 256, 0, stream>>>(pW2, pW2h, (long)512*512);
    f32_to_f16<<<2, 256, 0, stream>>>(pW3, pW3h, 512);

    // GAT layer 1
    gemm_f16<<<dim3(11, NR/64), 256, 0, stream>>>(node, g1Wh, nullptr, h1, nullptr,
                                                  (int)NR, 1344, 448, 448, 1344, 1344);
    att_logits<<<dim3(NMX, NB), 128, 0, stream>>>(h1, g1as, g1ad, as1, ad1, 448, 1344);
    att_rowstats<<<dim3(NMX, NB), 128, 0, stream>>>(as1, ad1, nat, rm1, rs1);
    gat_aggregate<<<dim3(7, 5, NB), 256, 0, stream>>>(h1, as1, ad1, rm1, rs1, nat, g1b,
                                                      nullptr, x1h, 448, 1344);
    // GAT layer 2
    gemm_f16<<<dim3(6, NR/64), 256, 0, stream>>>(x1h, g2Wh, nullptr, h2, nullptr,
                                                 (int)NR, 768, 448, 448, 768, 768);
    att_logits<<<dim3(NMX, NB), 128, 0, stream>>>(h2, g2as, g2ad, as2, ad2, 256, 768);
    att_rowstats<<<dim3(NMX, NB), 128, 0, stream>>>(as2, ad2, nat, rm2, rs2);
    gat_aggregate<<<dim3(4, 5, NB), 256, 0, stream>>>(h2, as2, ad2, rm2, rs2, nat, g2b,
                                                      x2f, nullptr, 256, 768);
    // readout + head
    sum_feat<<<NB, 256, 0, stream>>>(x2f, sgx, latx, feath);
    gemm_f16<<<dim3(4, NB/64), 256, 0, stream>>>(feath, pW1h, hb1, nullptr, pb1,
                                                 NB, 512, 448, 448, 512, 512);
    bn_lrelu<<<(NB*512)/256, 256, 0, stream>>>(hb1, pbn1g, pbn1b, pbn1m, pbn1v, hb1h,
                                               512, (long)NB*512);
    gemm_f16<<<dim3(4, NB/64), 256, 0, stream>>>(hb1h, pW2h, hb2, nullptr, pb2,
                                                 NB, 512, 512, 512, 512, 512);
    bn_lrelu<<<(NB*512)/256, 256, 0, stream>>>(hb2, pbn2g, pbn2b, pbn2m, pbn2v, hb2h,
                                               512, (long)NB*512);
    gemm_f16<<<dim3(1, NB/64), 256, 0, stream>>>(hb2h, pW3h, out, nullptr, pb3,
                                                 NB, 1, 512, 512, 1, 1);
}
